// GeniePathLazy_15917148799864
// MI455X (gfx1250) — matrix operand order repaired and checked by reading, not compiled, not run on hardware
//
#include <hip/hip_runtime.h>
#include <hip/hip_bf16.h>
#include <stdint.h>

// ---------------- problem constants ----------------
constexpr int   kN    = 100000;          // nodes
constexpr int   kE    = 1600000;         // edges (pre self-loop)
constexpr int   kEP   = kE + kN;         // edges + self loops
constexpr int   kDim  = 128;             // hidden dim
constexpr int   kOut  = 64;              // classes
constexpr int   kL    = 3;               // layers
constexpr int   kG    = 512;             // 4*DIM lstm gates
constexpr int   kKcat = 384;             // 3*DIM = [h_tmp | x | h]
constexpr float kNegSlope = 0.2f;
constexpr float kResW = 0.5f;

// ---------------- WMMA types ----------------
typedef __attribute__((ext_vector_type(16))) __bf16 v16bf;
typedef __attribute__((ext_vector_type(8)))  float  v8f;

union ABPack { uint4 q[2]; v16bf v; };

static __device__ __forceinline__ unsigned short f2bf(float f) {
  unsigned u = __float_as_uint(f);
  u += 0x7FFFu + ((u >> 16) & 1u);       // round to nearest even
  return (unsigned short)(u >> 16);
}
static __device__ __forceinline__ float sigf(float x) { return 1.0f / (1.0f + __expf(-x)); }

// ---------------- conversion / packing kernels ----------------
__global__ void k_f32_to_bf16(const float* __restrict__ in, unsigned short* __restrict__ out, int n) {
  int i = blockIdx.x * 256 + threadIdx.x;
  if (i < n) out[i] = f2bf(in[i]);
}

// Wcat[l][r][0:256] = Wih[l][r][:], Wcat[l][r][256:384] = Whh[l][r][:]
__global__ void k_build_wcat(const float* __restrict__ Wih, const float* __restrict__ Whh,
                             unsigned short* __restrict__ out) {
  int i = blockIdx.x * 256 + threadIdx.x;
  if (i >= kL * kG * kKcat) return;
  int l = i / (kG * kKcat);
  int rem = i - l * (kG * kKcat);
  int r = rem / kKcat;
  int k = rem - r * kKcat;
  float v = (k < 2 * kDim) ? Wih[(size_t)l * kG * 2 * kDim + (size_t)r * 2 * kDim + k]
                           : Whh[(size_t)l * kG * kDim + (size_t)r * kDim + (k - 2 * kDim)];
  out[i] = f2bf(v);
}

// Wt[l][j][k] = W[l][k][j]  (so xp = x @ W becomes B[n][k] contiguous in k)
__global__ void k_transpose_gat(const float* __restrict__ W, unsigned short* __restrict__ out) {
  int i = blockIdx.x * 256 + threadIdx.x;
  if (i >= kL * kDim * kDim) return;
  int l = i / (kDim * kDim);
  int rem = i - l * (kDim * kDim);
  int j = rem / kDim;
  int k = rem - j * kDim;
  out[i] = f2bf(W[(size_t)l * kDim * kDim + (size_t)k * kDim + j]);
}

__global__ void k_fill_u32(unsigned* __restrict__ p, unsigned v, int n) {
  int i = blockIdx.x * 256 + threadIdx.x;
  if (i < n) p[i] = v;
}

// ---------------- generic bf16 WMMA GEMM ----------------
// C[M,Nout] = A[M,K] (row-major, leading dim lda) x B[Nout,K]^T (+bias)
// One wave computes one 16x16 tile via v_wmma_f32_16x16x32_bf16.
__global__ __launch_bounds__(256) void k_gemm_bf16(
    const unsigned short* __restrict__ A, int lda,
    const unsigned short* __restrict__ B,
    const float* __restrict__ bias,
    float* __restrict__ outF,
    unsigned short* __restrict__ outH,
    int M, int Nout, int K) {
  const int warp = threadIdx.x >> 5;
  const int lane = threadIdx.x & 31;
  const int ntiles = Nout >> 4;
  const int t = blockIdx.x * 8 + warp;
  const int mt = t / ntiles;
  const int nt = t - mt * ntiles;
  if (mt >= (M >> 4)) return;
  const int hh = lane >> 4;            // half-wave select
  const int lr = lane & 15;
  const long m = (long)(mt * 16 + lr); // A row held by this lane (both halves)
  const long n = (long)(nt * 16 + lr); // B column held by this lane
  // A 16-bit 16x32 layout: lane half 0 -> K 0..7 & 16..23 ; half 1 -> +8
  const unsigned short* Ap = A + m * lda + hh * 8;
  // B 16-bit 32x16 layout: lane half 0 -> K 0..7 & 16..23 ; half 1 -> +8
  const unsigned short* Bp = B + n * K + hh * 8;
  v8f acc = {};
  for (int kk = 0; kk < K; kk += 32) {
    ABPack a, b;
    a.q[0] = *reinterpret_cast<const uint4*>(Ap + kk);        // K = 8h .. 8h+7
    a.q[1] = *reinterpret_cast<const uint4*>(Ap + kk + 16);   // K = 16+8h .. 16+8h+7
    b.q[0] = *reinterpret_cast<const uint4*>(Bp + kk);        // K = 8h .. 8h+7
    b.q[1] = *reinterpret_cast<const uint4*>(Bp + kk + 16);   // K = 16+8h .. 16+8h+7
    acc = __builtin_amdgcn_wmma_f32_16x16x32_bf16(false, a.v, false, b.v,
                                                  (short)0, acc, false, false);
  }
  // C/D layout: VGPR j -> row j (+8 for lanes>=16), col = lane%16
  const int col = nt * 16 + lr;
  const float bv = bias ? bias[col] : 0.0f;
#pragma unroll
  for (int j = 0; j < 8; ++j) {
    const int row = mt * 16 + j + hh * 8;
    const float v = acc[j] + bv;
    const long idx = (long)row * Nout + col;
    if (outF) outF[idx] = v;
    if (outH) outH[idx] = f2bf(v);
  }
}

// ---------------- GAT attention kernels ----------------
// one wave per node: alpha_src/alpha_dst dot products
__global__ void k_alpha(const float* __restrict__ xp, const float* __restrict__ av,
                        const float* __restrict__ dv, float* __restrict__ as_o,
                        float* __restrict__ ad_o) {
  const int w = (blockIdx.x * 256 + threadIdx.x) >> 5;
  const int lane = threadIdx.x & 31;
  if (w >= kN) return;
  const float4 xv = *reinterpret_cast<const float4*>(xp + (long)w * kDim + lane * 4);
  const float4 a4 = *reinterpret_cast<const float4*>(av + lane * 4);
  const float4 d4 = *reinterpret_cast<const float4*>(dv + lane * 4);
  float s = xv.x * a4.x + xv.y * a4.y + xv.z * a4.z + xv.w * a4.w;
  float d = xv.x * d4.x + xv.y * d4.y + xv.z * d4.z + xv.w * d4.w;
  for (int o = 16; o > 0; o >>= 1) { s += __shfl_xor(s, o, 32); d += __shfl_xor(d, o, 32); }
  if (lane == 0) { as_o[w] = s; ad_o[w] = d; }
}

// pass A: e = leaky_relu(as[src]+ad[dst]); segment-max via order-preserving u32 atomicMax
__global__ void k_edgeA(const int* __restrict__ ei, const float* __restrict__ as_,
                        const float* __restrict__ ad_, float* __restrict__ ebuf,
                        unsigned* __restrict__ menc) {
  const int e = blockIdx.x * 256 + threadIdx.x;
  if (e >= kEP) return;
  const int s = (e < kE) ? ei[e] : (e - kE);
  const int d = (e < kE) ? ei[kE + e] : (e - kE);
  float v = as_[s] + ad_[d];
  v = (v > 0.0f) ? v : kNegSlope * v;
  ebuf[e] = v;
  unsigned u = __float_as_uint(v);
  u = (u & 0x80000000u) ? ~u : (u | 0x80000000u);   // monotone encoding; 0 < encode(x) for all x
  atomicMax(menc + d, u);
}

// pass B: ex = exp(e - m[dst]); denom[dst] += ex
__global__ void k_edgeB(const int* __restrict__ ei, const float* __restrict__ ebuf,
                        const unsigned* __restrict__ menc, float* __restrict__ exbuf,
                        float* __restrict__ denom) {
  const int e = blockIdx.x * 256 + threadIdx.x;
  if (e >= kEP) return;
  const int d = (e < kE) ? ei[kE + e] : (e - kE);
  unsigned u = menc[d];
  u = (u & 0x80000000u) ? (u & 0x7FFFFFFFu) : ~u;
  const float mv = __uint_as_float(u);
  const float ex = __expf(ebuf[e] - mv);
  exbuf[e] = ex;
  atomicAdd(denom + d, ex);
}

// pass C: out[dst] += (ex/denom[dst]) * xp[src] ; one wave per edge, float4 per lane
__global__ void k_edgeC(const int* __restrict__ ei, const float* __restrict__ exbuf,
                        const float* __restrict__ denom, const float* __restrict__ xp,
                        float* __restrict__ accp) {
  const long g = ((long)blockIdx.x * 256 + threadIdx.x) >> 5;
  const int lane = threadIdx.x & 31;
  if (g >= kEP) return;
  const int e = (int)g;
  const int s = (e < kE) ? ei[e] : (e - kE);
  const int d = (e < kE) ? ei[kE + e] : (e - kE);
  const float coef = exbuf[e] / (denom[d] + 1e-16f);
  const float4 xv = *reinterpret_cast<const float4*>(xp + (long)s * kDim + lane * 4);
  float* op = accp + (long)d * kDim + lane * 4;
  atomicAdd(op + 0, coef * xv.x);
  atomicAdd(op + 1, coef * xv.y);
  atomicAdd(op + 2, coef * xv.z);
  atomicAdd(op + 3, coef * xv.w);
}

// h_tmp = tanh(acc + b)  (stored bf16)
__global__ void k_gat_finish(const float* __restrict__ acc, const float* __restrict__ b,
                             unsigned short* __restrict__ htmp) {
  const int i = blockIdx.x * 256 + threadIdx.x;
  if (i >= kN * kDim) return;
  htmp[i] = f2bf(tanhf(acc[i] + b[i & (kDim - 1)]));
}

// ---------------- LSTM input-assembly / pointwise ----------------
__global__ void k_init_acat(const unsigned short* __restrict__ x0h, unsigned short* __restrict__ acat) {
  const int i = blockIdx.x * 256 + threadIdx.x;
  if (i >= kN * kDim) return;
  const long ab = (long)(i >> 7) * kKcat;
  const int kcol = i & (kDim - 1);
  acat[ab + kDim + kcol] = x0h[i];   // x section = post-lin1 x
  acat[ab + 2 * kDim + kcol] = 0;    // h section = 0 (bf16 zero)
}

__global__ void k_copy_htmp(const unsigned short* __restrict__ src, unsigned short* __restrict__ acat) {
  const int i = blockIdx.x * 256 + threadIdx.x;
  if (i >= kN * kDim) return;
  acat[(long)(i >> 7) * kKcat + (i & (kDim - 1))] = src[i];
}

// gates order: i,f,g,o (torch). Updates c, writes next x/h bf16 directly into acat.
__global__ void k_lstm(const float* __restrict__ gates, const float* __restrict__ x0,
                       float* __restrict__ c, unsigned short* __restrict__ acat) {
  const int i = blockIdx.x * 256 + threadIdx.x;
  if (i >= kN * kDim) return;
  const int nrow = i >> 7;
  const int kcol = i & (kDim - 1);
  const long gb = (long)nrow * kG;
  const float ii = gates[gb + kcol];
  const float ff = gates[gb + kDim + kcol];
  const float gg = gates[gb + 2 * kDim + kcol];
  const float oo = gates[gb + 3 * kDim + kcol];
  const float cv = sigf(ff) * c[i] + sigf(ii) * tanhf(gg);
  const float hv = sigf(oo) * tanhf(cv);
  c[i] = cv;
  const float xn = hv + kResW * x0[i];
  const long ab = (long)nrow * kKcat;
  acat[ab + kDim + kcol] = f2bf(xn);
  acat[ab + 2 * kDim + kcol] = f2bf(hv);
}

// in-place log_softmax over 64 columns; one wave per row, 2 cols per lane
__global__ void k_log_softmax64(float* __restrict__ out) {
  const long w = ((long)blockIdx.x * 256 + threadIdx.x) >> 5;
  const int lane = threadIdx.x & 31;
  if (w >= kN) return;
  float* row = out + w * kOut;
  const float v0 = row[lane];
  const float v1 = row[lane + 32];
  float m = fmaxf(v0, v1);
  for (int o = 16; o > 0; o >>= 1) m = fmaxf(m, __shfl_xor(m, o, 32));
  float s = __expf(v0 - m) + __expf(v1 - m);
  for (int o = 16; o > 0; o >>= 1) s += __shfl_xor(s, o, 32);
  const float ls = m + __logf(s);
  row[lane] = v0 - ls;
  row[lane + 32] = v1 - ls;
}

// ---------------- launch ----------------
extern "C" void kernel_launch(void* const* d_in, const int* in_sizes, int n_in,
                              void* d_out, int out_size, void* d_ws, size_t ws_size,
                              hipStream_t stream) {
  (void)in_sizes; (void)n_in; (void)out_size; (void)ws_size;
  const float* x      = (const float*)d_in[0];
  const int*   ei     = (const int*)d_in[1];
  const float* lin1_w = (const float*)d_in[2];
  const float* lin1_b = (const float*)d_in[3];
  const float* gat_W  = (const float*)d_in[4];
  const float* att_s  = (const float*)d_in[5];
  const float* att_d  = (const float*)d_in[6];
  const float* gat_b  = (const float*)d_in[7];
  const float* Wih    = (const float*)d_in[8];
  const float* Whh    = (const float*)d_in[9];
  const float* lin2_w = (const float*)d_in[10];
  const float* lin2_b = (const float*)d_in[11];
  float* out = (float*)d_out;

  char* base = (char*)d_ws;
  size_t off = 0;
  auto alloc = [&](size_t bytes) -> char* {
    char* p = base + off;
    off += (bytes + 255) & ~(size_t)255;
    return p;
  };
  unsigned short* xin_h  = (unsigned short*)alloc((size_t)kN * kDim * 2);
  unsigned short* w1_h   = (unsigned short*)alloc((size_t)kDim * kDim * 2);
  unsigned short* wgt_h  = (unsigned short*)alloc((size_t)kL * kDim * kDim * 2);
  unsigned short* wcat_h = (unsigned short*)alloc((size_t)kL * kG * kKcat * 2);
  unsigned short* w2_h   = (unsigned short*)alloc((size_t)kOut * kDim * 2);
  float*          x0f    = (float*)alloc((size_t)kN * kDim * 4);
  unsigned short* x0h    = (unsigned short*)alloc((size_t)kN * kDim * 2);
  unsigned short* htmp   = (unsigned short*)alloc((size_t)kL * kN * kDim * 2);
  float*          xp     = (float*)alloc((size_t)kN * kDim * 4);
  float*          accp   = (float*)alloc((size_t)kN * kDim * 4);
  float*          asrc   = (float*)alloc((size_t)kN * 4);
  float*          adst   = (float*)alloc((size_t)kN * 4);
  unsigned*       menc   = (unsigned*)alloc((size_t)kN * 4);
  float*          denom  = (float*)alloc((size_t)kN * 4);
  float*          ebuf   = (float*)alloc((size_t)kEP * 4);
  float*          exbuf  = (float*)alloc((size_t)kEP * 4);
  unsigned short* acat   = (unsigned short*)alloc((size_t)kN * kKcat * 2);
  float*          gates  = (float*)alloc((size_t)kN * kG * 4);
  float*          cbuf   = (float*)alloc((size_t)kN * kDim * 4);

  const int NK = kN * kDim;                       // 12.8M, multiple of 256

  // weight / input packing (bf16)
  k_f32_to_bf16<<<NK / 256, 256, 0, stream>>>(x, xin_h, NK);
  k_f32_to_bf16<<<(kDim * kDim) / 256, 256, 0, stream>>>(lin1_w, w1_h, kDim * kDim);
  k_transpose_gat<<<(kL * kDim * kDim) / 256, 256, 0, stream>>>(gat_W, wgt_h);
  k_build_wcat<<<(kL * kG * kKcat) / 256, 256, 0, stream>>>(Wih, Whh, wcat_h);
  k_f32_to_bf16<<<(kOut * kDim) / 256, 256, 0, stream>>>(lin2_w, w2_h, kOut * kDim);
  k_fill_u32<<<NK / 256, 256, 0, stream>>>((unsigned*)cbuf, 0u, NK);

  // lin1: x0 = x @ lin1_w^T + b (f32 + bf16 copies)
  k_gemm_bf16<<<(kN / 16) * (kDim / 16) / 8, 256, 0, stream>>>(
      xin_h, kDim, w1_h, lin1_b, x0f, x0h, kN, kDim, kDim);
  k_init_acat<<<NK / 256, 256, 0, stream>>>(x0h, acat);

  // all three GAT layers read the same x0
  for (int l = 0; l < kL; ++l) {
    k_gemm_bf16<<<(kN / 16) * (kDim / 16) / 8, 256, 0, stream>>>(
        x0h, kDim, wgt_h + (size_t)l * kDim * kDim, nullptr, xp, nullptr, kN, kDim, kDim);
    k_alpha<<<kN * 32 / 256, 256, 0, stream>>>(xp, att_s + l * kDim, att_d + l * kDim, asrc, adst);
    k_fill_u32<<<(kN + 255) / 256, 256, 0, stream>>>(menc, 0u, kN);
    k_fill_u32<<<(kN + 255) / 256, 256, 0, stream>>>((unsigned*)denom, 0u, kN);
    k_fill_u32<<<NK / 256, 256, 0, stream>>>((unsigned*)accp, 0u, NK);
    k_edgeA<<<(kEP + 255) / 256, 256, 0, stream>>>(ei, asrc, adst, ebuf, menc);
    k_edgeB<<<(kEP + 255) / 256, 256, 0, stream>>>(ei, ebuf, menc, exbuf, denom);
    k_edgeC<<<kEP * 32 / 256, 256, 0, stream>>>(ei, exbuf, denom, xp, accp);
    k_gat_finish<<<NK / 256, 256, 0, stream>>>(accp, gat_b + l * kDim, htmp + (size_t)l * NK);
  }

  // LSTM recurrence
  for (int l = 0; l < kL; ++l) {
    k_copy_htmp<<<NK / 256, 256, 0, stream>>>(htmp + (size_t)l * NK, acat);
    k_gemm_bf16<<<(kN / 16) * (kG / 16) / 8, 256, 0, stream>>>(
        acat, kKcat, wcat_h + (size_t)l * kG * kKcat, nullptr, gates, nullptr, kN, kG, kKcat);
    k_lstm<<<NK / 256, 256, 0, stream>>>(gates, x0f, cbuf, acat);
  }

  // lin2 (+ bias) straight into d_out, then in-place log_softmax
  k_gemm_bf16<<<(kN / 16) * (kOut / 16) / 8, 256, 0, stream>>>(
      acat + kDim, kKcat, w2_h, lin2_b, out, nullptr, kN, kOut, kDim);
  k_log_softmax64<<<kN * 32 / 256, 256, 0, stream>>>(out);
}
